// LinearAttention_81432579932229
// MI455X (gfx1250) — matrix operand order repaired and checked by reading, not compiled, not run on hardware
//
#include <hip/hip_runtime.h>
#include <math.h>

#define SEQ    4096
#define BATCH  4
#define CM     1024
#define NHEADS 16
#define HDIM   64
#define EPSF   1e-6f

typedef __attribute__((ext_vector_type(16))) __bf16 v16bf;
typedef __attribute__((ext_vector_type(8)))  float  v8f;
typedef unsigned short u16;
typedef unsigned int   u32;

union FragU { uint4 q[2]; v16bf v; };

__device__ __forceinline__ u32 pack2_bf16(float lo, float hi) {
    u32 a = __float_as_uint(lo);
    u32 b = __float_as_uint(hi);
    a = (a + 0x7FFFu + ((a >> 16) & 1u)) >> 16;      // RNE
    b = (b + 0x7FFFu + ((b >> 16) & 1u)) & 0xFFFF0000u;
    return (a & 0xFFFFu) | b;
}
__device__ __forceinline__ u16 f32_to_bf16(float f) {
    u32 u = __float_as_uint(f);
    return (u16)((u + 0x7FFFu + ((u >> 16) & 1u)) >> 16);
}
__device__ __forceinline__ float bf16_to_f32(u16 h) {
    return __uint_as_float(((u32)h) << 16);
}

// ---------------- WMMA fragment loaders (bf16 16x16x32) ------------------
// A (16xK, row-major, K contiguous): lane<16 -> M=lane, K {0..7,16..23};
// lane>=16 -> M=lane-16, K {8..15,24..31}. Two b128 loads per lane.
__device__ __forceinline__ v16bf load_fragA(const u16* __restrict__ p0, int ld) {
    int lane = threadIdx.x & 31;
    const u16* p = p0 + (lane & 15) * ld + ((lane >> 4) << 3);
    FragU u;
    u.q[0] = *(const uint4*)(p);
    u.q[1] = *(const uint4*)(p + 16);
    return u.v;
}
// B (KxN) loaded from Bt = B^T row-major (N rows, K contiguous):
// lane<16 -> N=lane, K {0..7,16..23} ; lane>=16 -> N=lane-16, K {8..15,24..31}.
__device__ __forceinline__ v16bf load_fragB(const u16* __restrict__ p0, int ld) {
    int lane = threadIdx.x & 31;
    const u16* p = p0 + (lane & 15) * ld + ((lane >> 4) << 3);
    FragU u;
    u.q[0] = *(const uint4*)(p);
    u.q[1] = *(const uint4*)(p + 16);
    return u.v;
}
__device__ __forceinline__ v8f wmma_bf16(v16bf a, v16bf b, v8f c) {
    return __builtin_amdgcn_wmma_f32_16x16x32_bf16(false, a, false, b, (short)0, c, false, false);
}

// ---------------- Stage 1: fp32 -> bf16, 8 elements / thread --------------
__global__ void cvt_f32_bf16_kernel(const float4* __restrict__ src, uint4* __restrict__ dst, int n8) {
    int i = blockIdx.x * blockDim.x + threadIdx.x;
    int stride = gridDim.x * blockDim.x;
    for (; i < n8; i += stride) {
        float4 a = src[2 * i];
        float4 b = src[2 * i + 1];
        uint4 o;
        o.x = pack2_bf16(a.x, a.y);
        o.y = pack2_bf16(a.z, a.w);
        o.z = pack2_bf16(b.x, b.y);
        o.w = pack2_bf16(b.z, b.w);
        dst[i] = o;
    }
}

// ---------------- Stages 2 & 5: projection GEMM ---------------------------
// OUT[row,col] = sum_k A[row,k]*W[col,k] + bias[col], M=BATCH*SEQ, Kd=N=CM.
// Per wave: 64x32 tile (8 WMMAs / k-step, ~170 FLOP per L2 byte), 8 waves
// -> 128x128 block tile. __launch_bounds__(256,1) lifts the VGPR cap so the
// 8 f32 accumulators + pipelined fragments stay in registers (no scratch).
// mode 0: q  -> elu+1, bf16, [B,N,C]
// mode 1: k  -> elu+1, bf16, transposed to [B,H,D,N]
// mode 2: v  -> bf16, transposed to [B,H,D,N]
// mode 3: f32 out, [B,N,C]
__global__ void __launch_bounds__(256, 1)
gemm_proj_kernel(const u16* __restrict__ Abf, const u16* __restrict__ Wbf,
                 const float* __restrict__ bias,
                 u16* __restrict__ dstb, float* __restrict__ dstf, int mode) {
    int wave = threadIdx.x >> 5;
    int row0 = blockIdx.x * 128 + (wave >> 2) * 64;
    int col0 = blockIdx.y * 128 + (wave & 3) * 32;
    const u16* a0 = Abf + (size_t)row0 * CM;
    const u16* b0 = Wbf + (size_t)col0 * CM;

    v8f acc[4][2] = {};
    for (int k = 0; k < CM; k += 32) {
        v16bf af[4], bf[2];
#pragma unroll
        for (int i = 0; i < 4; ++i) af[i] = load_fragA(a0 + (size_t)i * 16 * CM + k, CM);
#pragma unroll
        for (int j = 0; j < 2; ++j) bf[j] = load_fragB(b0 + (size_t)j * 16 * CM + k, CM);
#pragma unroll
        for (int i = 0; i < 4; ++i)
#pragma unroll
            for (int j = 0; j < 2; ++j)
                acc[i][j] = wmma_bf16(af[i], bf[j], acc[i][j]);
    }

    int lane  = threadIdx.x & 31;
    int cn    = lane & 15;
    int rbase = (lane >> 4) * 8;
#pragma unroll
    for (int ti = 0; ti < 4; ++ti)
#pragma unroll
    for (int tj = 0; tj < 2; ++tj) {
        int col = col0 + tj * 16 + cn;
        float bcol = bias[col];
        if (mode == 1 || mode == 2) {
            // transposed store [B,H,D,N]: 8 rows are contiguous in n -> one b128
            int rowb = row0 + ti * 16 + rbase;            // multiple of 8
            int b = rowb >> 12, nn0 = rowb & (SEQ - 1);
            int h = col >> 6,   d   = col & (HDIM - 1);
            float vv[8];
#pragma unroll
            for (int r = 0; r < 8; ++r) {
                float t = acc[ti][tj][r] + bcol;
                vv[r] = (mode == 1) ? ((t > 0.f) ? (t + 1.f) : __expf(t)) : t;
            }
            uint4 o;
            o.x = pack2_bf16(vv[0], vv[1]);
            o.y = pack2_bf16(vv[2], vv[3]);
            o.z = pack2_bf16(vv[4], vv[5]);
            o.w = pack2_bf16(vv[6], vv[7]);
            *(uint4*)(dstb + (((size_t)b * NHEADS + h) * HDIM + d) * SEQ + nn0) = o;
        } else {
#pragma unroll
            for (int r = 0; r < 8; ++r) {
                int row = row0 + ti * 16 + rbase + r;
                float t = acc[ti][tj][r] + bcol;
                if (mode == 0) {
                    t = (t > 0.f) ? (t + 1.f) : __expf(t);   // elu(x)+1
                    dstb[(size_t)row * CM + col] = f32_to_bf16(t);
                } else {
                    dstf[(size_t)row * CM + col] = t;
                }
            }
        }
    }
}

// ---------------- Stage 3: kv = K^T @ V (per b,h) + z row-sums ------------
// kT, vT: [B,H,D,N] (n contiguous). kvT out: [B,H, m, d] bf16. z: [B,H,D] f32.
__global__ void kv_kernel(const u16* __restrict__ kT, const u16* __restrict__ vT,
                          u16* __restrict__ kvT, float* __restrict__ z) {
    int bh = blockIdx.x;                       // 0..63
    const u16* kh = kT + (size_t)bh * HDIM * SEQ;
    const u16* vh = vT + (size_t)bh * HDIM * SEQ;

    int wave = threadIdx.x >> 5;
    int mi   = wave >> 1;                      // d-tile 0..3
    int njb  = (wave & 1) * 2;                 // m-tiles {0,1} or {2,3}

    v8f acc[2] = {};
    for (int k = 0; k < SEQ; k += 32) {
        v16bf a  = load_fragA(kh + (size_t)mi * 16 * SEQ + k, SEQ);
        v16bf b0 = load_fragB(vh + (size_t)njb * 16 * SEQ + k, SEQ);
        v16bf b1 = load_fragB(vh + (size_t)(njb + 1) * 16 * SEQ + k, SEQ);
        acc[0] = wmma_bf16(a, b0, acc[0]);
        acc[1] = wmma_bf16(a, b1, acc[1]);
    }
    int lane  = threadIdx.x & 31;
    int cn    = lane & 15;
    int rbase = (lane >> 4) * 8;
#pragma unroll
    for (int t = 0; t < 2; ++t)
#pragma unroll
    for (int r = 0; r < 8; ++r) {
        int d = mi * 16 + rbase + r;
        int m = (njb + t) * 16 + cn;
        kvT[(size_t)bh * HDIM * HDIM + m * HDIM + d] = f32_to_bf16(acc[t][r]);
    }

    // z[d] = sum_n k[d][n]
    __shared__ float zs[HDIM];
    if (threadIdx.x < HDIM) zs[threadIdx.x] = 0.f;
    __syncthreads();
    int d   = threadIdx.x & (HDIM - 1);
    int seg = threadIdx.x >> 6;                // 0..3
    const u16* kp = kh + (size_t)d * SEQ + seg * (SEQ / 4);
    float s = 0.f;
    for (int i = 0; i < SEQ / 4; ++i) s += bf16_to_f32(kp[i]);
    atomicAdd(&zs[d], s);
    __syncthreads();
    if (threadIdx.x < HDIM) z[(size_t)bh * HDIM + threadIdx.x] = zs[threadIdx.x];
}

// ---------------- Stage 4: out = (q @ kv) / (q.z + eps) -------------------
__global__ void attn_kernel(const u16* __restrict__ q, const u16* __restrict__ kvT,
                            const float* __restrict__ z, u16* __restrict__ attn) {
    int b = blockIdx.y, h = blockIdx.z;
    int wave = threadIdx.x >> 5;
    int tile = blockIdx.x * 8 + wave;          // 16-row tile in sequence
    int n0   = tile * 16;

    const u16*  qb = q   + ((size_t)(b * SEQ + n0)) * CM + h * HDIM;
    const u16*  kv = kvT + (size_t)(b * NHEADS + h) * HDIM * HDIM;
    const float* zh = z  + (size_t)(b * NHEADS + h) * HDIM;

    v8f acc[4] = {};
#pragma unroll
    for (int ks = 0; ks < 2; ++ks) {
        v16bf a = load_fragA(qb + ks * 32, CM);
#pragma unroll
        for (int nj = 0; nj < 4; ++nj) {
            v16bf bb = load_fragB(kv + nj * 16 * HDIM + ks * 32, HDIM);
            acc[nj] = wmma_bf16(a, bb, acc[nj]);
        }
    }

    int lane = threadIdx.x & 31;
    // lane L computes denom of row (L&15); lanes 16..31 duplicate
    float denom = EPSF;
    {
        const u16* qr = qb + (size_t)(lane & 15) * CM;
        for (int dd = 0; dd < HDIM; ++dd) denom += bf16_to_f32(qr[dd]) * zh[dd];
    }
    int cn    = lane & 15;
    int rbase = (lane >> 4) * 8;
#pragma unroll
    for (int nj = 0; nj < 4; ++nj)
#pragma unroll
    for (int r = 0; r < 8; ++r) {
        float dj  = __shfl(denom, rbase + r, 32);
        int   row = n0 + rbase + r;
        int   col = h * HDIM + nj * 16 + cn;
        attn[((size_t)(b * SEQ + row)) * CM + col] = f32_to_bf16(acc[nj][r] / dj);
    }
}

// --------------------------------------------------------------------------
extern "C" void kernel_launch(void* const* d_in, const int* in_sizes, int n_in,
                              void* d_out, int out_size, void* d_ws, size_t ws_size,
                              hipStream_t stream) {
    const float* x  = (const float*)d_in[0];
    const float* Wq = (const float*)d_in[1];
    const float* bq = (const float*)d_in[2];
    const float* Wk = (const float*)d_in[3];
    const float* bk = (const float*)d_in[4];
    const float* Wv = (const float*)d_in[5];
    const float* bv = (const float*)d_in[6];
    const float* Wo = (const float*)d_in[7];
    const float* bo = (const float*)d_in[8];
    float* out = (float*)d_out;

    const size_t NX = (size_t)BATCH * SEQ * CM;     // 16,777,216
    const size_t NW = (size_t)CM * CM;              // 1,048,576

    char* ws = (char*)d_ws;
    size_t off = 0;
    u16* xb  = (u16*)(ws + off); off += NX * 2;
    u16* wqb = (u16*)(ws + off); off += NW * 2;
    u16* wkb = (u16*)(ws + off); off += NW * 2;
    u16* wvb = (u16*)(ws + off); off += NW * 2;
    u16* wob = (u16*)(ws + off); off += NW * 2;
    u16* qb  = (u16*)(ws + off); off += NX * 2;     // [B,N,C]
    u16* kTb = (u16*)(ws + off); off += NX * 2;     // [B,H,D,N]
    u16* vTb = (u16*)(ws + off); off += NX * 2;     // [B,H,D,N]
    u16* kvb = (u16*)(ws + off); off += (size_t)BATCH * NHEADS * HDIM * HDIM * 2;
    float* zf = (float*)(ws + off); off += (size_t)BATCH * NHEADS * HDIM * 4;
    u16* atb = (u16*)(ws + off); off += NX * 2;     // [B,N,C]
    (void)ws_size; (void)n_in; (void)in_sizes; (void)out_size;

    // 1) fp32 -> bf16 (vectorized: 8 elems/thread)
    cvt_f32_bf16_kernel<<<2048, 256, 0, stream>>>((const float4*)x,  (uint4*)xb,  (int)(NX / 8));
    cvt_f32_bf16_kernel<<<512,  256, 0, stream>>>((const float4*)Wq, (uint4*)wqb, (int)(NW / 8));
    cvt_f32_bf16_kernel<<<512,  256, 0, stream>>>((const float4*)Wk, (uint4*)wkb, (int)(NW / 8));
    cvt_f32_bf16_kernel<<<512,  256, 0, stream>>>((const float4*)Wv, (uint4*)wvb, (int)(NW / 8));
    cvt_f32_bf16_kernel<<<512,  256, 0, stream>>>((const float4*)Wo, (uint4*)wob, (int)(NW / 8));

    // 2) QKV projections (WMMA, 64x32 per wave, 128x128 per block)
    dim3 gp((BATCH * SEQ) / 128, CM / 128, 1);
    gemm_proj_kernel<<<gp, 256, 0, stream>>>(xb, wqb, bq, qb,  nullptr, 0);
    gemm_proj_kernel<<<gp, 256, 0, stream>>>(xb, wkb, bk, kTb, nullptr, 1);
    gemm_proj_kernel<<<gp, 256, 0, stream>>>(xb, wvb, bv, vTb, nullptr, 2);

    // 3) kv + z per head (WMMA)
    kv_kernel<<<BATCH * NHEADS, 256, 0, stream>>>(kTb, vTb, kvb, zf);

    // 4) normalized attention output (WMMA)
    dim3 ga(SEQ / 128, BATCH, NHEADS);
    attn_kernel<<<ga, 256, 0, stream>>>(qb, kvb, zf, atb);

    // 5) output projection (WMMA, fp32 out)
    gemm_proj_kernel<<<gp, 256, 0, stream>>>(atb, wob, bo, nullptr, out, 3);
}
